// VRNN_39479339385364
// MI455X (gfx1250) — compile-verified, matrix operand order checked by reading, not run on hardware
//
#include <hip/hip_runtime.h>
#include <hip/hip_bf16.h>

// ---------------------------------------------------------------------------
// VRNN (H=256, Z=X=5, T=3000) for MI455X / gfx1250, wave32 + WMMA bf16.
//   Phase A (parallel): precompute phi_x, enc_x-part, gru_ih x-part for all t
//                       as real bf16 WMMA GEMMs (M=3008 padded, K=256).
//   Phase B (sequential): single persistent 1024-thread workgroup runs the
//                       recurrence; H x H GEMVs done as WMMA with broadcast B.
//                       enc_W2/dec_W2 tiles staged in LDS (256 KB of the WGP's
//                       320 KB) so the under-occupied phases avoid L2 latency;
//                       next-step activations prefetched via global_prefetch.
// ---------------------------------------------------------------------------

#define H   256
#define Zd  5
#define Xd  5
#define T   3000
#define TP  3008            // padded to 188 * 16 rows
#define MT  188             // M tiles in precompute GEMMs
#define EPSF 1e-9f

typedef __attribute__((ext_vector_type(16))) __bf16 v16bf;
typedef __attribute__((ext_vector_type(8)))  float  v8f;

union BV { v16bf v; uint4 q[2]; };

__device__ inline unsigned short f2bf(float f) {
  union { float f; unsigned u; } a; a.f = f;
  unsigned u = a.u;
  u += 0x7fffu + ((u >> 16) & 1u);       // round-to-nearest-even
  return (unsigned short)(u >> 16);
}

// 16 contiguous bf16 (tiled-weight layout): one 32B fetch per lane
__device__ inline v16bf load_bf_chunk(const unsigned short* p) {
  BV b; b.q[0] = *(const uint4*)p; b.q[1] = *(const uint4*)(p + 8); return b.v;
}
// Two 8-element runs at p[0..7] and p[16..23] (row-major A / broadcast B)
__device__ inline v16bf load_bf_split(const unsigned short* p) {
  BV b; b.q[0] = *(const uint4*)p; b.q[1] = *(const uint4*)(p + 16); return b.v;
}

__device__ inline v8f bwmma(v16bf a, v16bf b, v8f c) {
  return __builtin_amdgcn_wmma_f32_16x16x32_bf16(false, a, false, b, (short)0, c,
                                                 false, false);
}

__device__ inline float softplusf(float v) {
  return (v > 20.f) ? v : log1pf(expf(v));
}
__device__ inline float sigmoidf(float v) { return 1.f / (1.f + expf(-v)); }

// GEMV out[16] = Wtile(16x256) @ vec(256), weights in tiled bf16 layout
// (tile = 8 chunks of 512 bf16; chunk element (lane*16+e)).  Wt may point to
// global (L2-resident) or LDS-staged tiles; vb is an LDS bf16 vector.
__device__ inline void gemv_tile(const unsigned short* Wt,
                                 const unsigned short* vb,  // LDS bf16 vec[256]
                                 int lane, float* dst) {
  v8f c = {};
  const int klo = (lane & 16) >> 1;      // 0 or 8
#pragma unroll
  for (int kc = 0; kc < 8; ++kc) {
    v16bf a = load_bf_chunk(Wt + kc * 512 + lane * 16);
    v16bf b = load_bf_split(vb + kc * 32 + klo);   // broadcast columns
    c = bwmma(a, b, c);
  }
  if ((lane & 15) == 0) {                // lane 0 -> rows 0..7, lane 16 -> 8..15
    const int rbase = (lane >> 4) << 3;
#pragma unroll
    for (int r = 0; r < 8; ++r) dst[rbase + r] = c[r];
  }
}

// ---------------------------------------------------------------------------
// Convert fp32 weight (N x ldW, column window colOff..colOff+255) into the
// WMMA tile layout: out[(nt*8+kc)*512 + lane*16 + e] = W[nt*16 + (lane&15)]
//                   [colOff + kc*32 + e + (e&8) + ((lane&16)>>1)]
__global__ void k_tile(const float* __restrict__ W, int ldW, int colOff,
                       unsigned short* __restrict__ out) {
  const int idx = blockIdx.x;            // nt*8 + kc
  const int nt = idx >> 3, kc = idx & 7;
  for (int i = threadIdx.x; i < 512; i += 256) {
    const int lane = i >> 4, e = i & 15;
    const int k = kc * 32 + ((lane & 16) >> 1) + e + (e & 8);
    const int m = lane & 15;
    out[idx * 512 + i] = f2bf(W[(nt * 16 + m) * ldW + colOff + k]);
  }
}

// Layer 1 of phi_x (K=5, VALU): tmp1 = relu(x @ W1^T + b1) -> bf16, padded.
__global__ void k_layer1(const float* __restrict__ x, const float* __restrict__ W1,
                         const float* __restrict__ b1,
                         unsigned short* __restrict__ out) {
  const int t = blockIdx.x, i = threadIdx.x;
  float f = 0.f;
  if (t < T) {
    f = b1[i];
#pragma unroll
    for (int j = 0; j < Xd; ++j) f += W1[i * Xd + j] * x[t * Xd + j];
    f = fmaxf(f, 0.f);
  }
  out[t * H + i] = f2bf(f);
}

// ---------------------------------------------------------------------------
// Generic bf16 WMMA GEMM: out(M x N) = A(M x 256) @ W^T, W given tiled.
struct GemmArgs {
  const unsigned short* A;     // row-major bf16, ld = 256, M = TP
  const unsigned short* Bt;    // tiled weights, ntiles n-tiles
  const float* bias;           // length ntiles*16 or null
  float* outF;                 // row-major, ld = Nld (or null)
  unsigned short* outBf;       // row-major bf16, ld = 256 (or null)
  int ntiles, Nld, relu;
};

__global__ void __launch_bounds__(256) k_gemm(GemmArgs g) {
  const int lane = threadIdx.x & 31;
  const int wave = threadIdx.x >> 5;
  const int tile = blockIdx.x * 8 + wave;
  const int mt = tile / g.ntiles, nt = tile % g.ntiles;
  const int klo = (lane & 16) >> 1;
  const unsigned short* arow = g.A + (mt * 16 + (lane & 15)) * 256;
  const unsigned short* bt = g.Bt + nt * 4096;
  v8f c = {};
#pragma unroll
  for (int kc = 0; kc < 8; ++kc) {
    v16bf a = load_bf_split(arow + kc * 32 + klo);
    v16bf b = load_bf_chunk(bt + kc * 512 + lane * 16);
    c = bwmma(a, b, c);
  }
  const int n = nt * 16 + (lane & 15);
  const int rbase = mt * 16 + ((lane >> 4) << 3);
  const float bv = g.bias ? g.bias[n] : 0.f;
#pragma unroll
  for (int r = 0; r < 8; ++r) {
    float f = c[r] + bv;
    if (g.relu) f = fmaxf(f, 0.f);
    if (g.outF)  g.outF[(rbase + r) * g.Nld + n] = f;
    if (g.outBf) g.outBf[(rbase + r) * 256 + n] = f2bf(f);
  }
}

// ---------------------------------------------------------------------------
struct RecArgs {
  const float *x, *eps;
  const unsigned short *G1, *G2, *G3, *G4;   // tiled recurrent weights
  const float *encx, *gix;                   // precomputed (TP x H), (TP x 3H)
  const float *prior_b, *enc_b2, *dec_b1, *dec_b2;
  const float *phi_z_W, *phi_z_b;
  const float *enc_mean_W, *enc_mean_b, *enc_std_W, *enc_std_b;
  const float *prior_mean_W, *prior_mean_b, *prior_std_W, *prior_std_b;
  const float *dec_mean_W, *dec_mean_b, *dec_std_W, *dec_std_b;
  float* out;
};

__global__ void __launch_bounds__(1024) k_recurrent(RecArgs A) {
  // Dynamic LDS: 32 WMMA tiles (enc_W2: 0..15, dec_W2: 16..31) = 256 KB.
  extern __shared__ unsigned short ldsW[];
  __shared__ __align__(16) float h[H], hn[H], u_ench[H], prior_h[H], u_dec1h[H];
  __shared__ __align__(16) float enc_h[H], dec_h[H];
  __shared__ __align__(16) float gh[3 * H], giz[3 * H];
  __shared__ __align__(16) unsigned short hb[H], enc1b[H], phizb[H], dec1b[H];
  __shared__ float em[8], es[8], pm[8], ps[8], zv[8], dm[8];
  __shared__ float acc[2];

  const int tid = threadIdx.x, lane = tid & 31, wave = tid >> 5;

  // Stage enc_W2 / dec_W2 tiles into LDS (128-bit copies, fully coalesced).
  {
    const uint4* s2 = (const uint4*)A.G2;
    const uint4* s4 = (const uint4*)A.G4;
    uint4* dW = (uint4*)ldsW;
    const int n = 16 * 4096 / 8;           // 8192 uint4 per 16-tile group
    for (int i = tid; i < n; i += 1024) dW[i] = s2[i];
    for (int i = tid; i < n; i += 1024) dW[n + i] = s4[i];
  }
  if (tid < H) { h[tid] = 0.f; hb[tid] = 0; }
  if (tid < 2) acc[tid] = 0.f;
  __syncthreads();

  for (int t = 0; t < T; ++t) {
    // P1: all GEMVs on h: enc_W1h(16) | prior(16) | dec_W1h(16) | Whh(48) tiles
#pragma unroll
    for (int i = 0; i < 3; ++i) {
      const int tt = wave * 3 + i;
      float* dst;
      if (tt < 16)      dst = u_ench + tt * 16;
      else if (tt < 32) dst = prior_h + (tt - 16) * 16;
      else if (tt < 48) dst = u_dec1h + (tt - 32) * 16;
      else              dst = gh + (tt - 48) * 16;
      gemv_tile(A.G1 + tt * 4096, hb, lane, dst);
    }
    __syncthreads();
    // P2: enc layer1 combine (+precomputed x-part) ; prior_h activation
    if (tid < H) {
      enc1b[tid] = f2bf(fmaxf(A.encx[t * H + tid] + u_ench[tid], 0.f));
      prior_h[tid] = fmaxf(prior_h[tid] + A.prior_b[tid], 0.f);
    }
    // Prefetch next step's activation rows into near caches (idle lanes).
    if (t + 1 < T && tid >= 512 && tid < 544) {
      const int i = tid - 512;             // 0..31
      if (i < 8) __builtin_prefetch(A.encx + (t + 1) * H + i * 32, 0, 3);
      else       __builtin_prefetch(A.gix + (t + 1) * 768 + (i - 8) * 32, 0, 3);
    }
    __syncthreads();
    // P3: enc_h = enc_W2 @ enc1 (waves 0-15, LDS tiles) ; prior mean/std (16-25)
    if (wave < 16) {
      gemv_tile(ldsW + wave * 4096, enc1b, lane, enc_h + wave * 16);
    } else if (wave < 26) {
      const int r = wave - 16, rr = (r < 5) ? r : r - 5;
      const float* Wm = (r < 5) ? A.prior_mean_W : A.prior_std_W;
      float s = 0.f;
      for (int j = lane; j < H; j += 32) s += Wm[rr * H + j] * prior_h[j];
      for (int off = 16; off; off >>= 1) s += __shfl_xor(s, off, 32);
      if (lane == 0) {
        if (r < 5) pm[rr] = s + A.prior_mean_b[rr];
        else       ps[rr] = softplusf(s + A.prior_std_b[rr]);
      }
    }
    __syncthreads();
    // P4: enc_h activation
    if (tid < H) enc_h[tid] = fmaxf(enc_h[tid] + A.enc_b2[tid], 0.f);
    __syncthreads();
    // P5a: enc mean/std (waves 0-9)
    if (wave < 10) {
      const int r = wave, rr = (r < 5) ? r : r - 5;
      const float* Wm = (r < 5) ? A.enc_mean_W : A.enc_std_W;
      float s = 0.f;
      for (int j = lane; j < H; j += 32) s += Wm[rr * H + j] * enc_h[j];
      for (int off = 16; off; off >>= 1) s += __shfl_xor(s, off, 32);
      if (lane == 0) {
        if (r < 5) em[rr] = s + A.enc_mean_b[rr];
        else       es[rr] = softplusf(s + A.enc_std_b[rr]);
      }
    }
    __syncthreads();
    // P5b: reparameterize z, emit enc outputs
    if (tid < Zd) {
      const float zz = A.eps[t * Zd + tid] * es[tid] + em[tid];
      zv[tid] = zz;
      A.out[2 + t * Zd + tid]              = em[tid];
      A.out[2 + T * Zd + t * Zd + tid]     = es[tid];
      A.out[2 + 4 * T * Zd + t * Zd + tid] = zz;
    }
    __syncthreads();
    // P5c: phi_z (K=5 VALU) ; KLD accumulation
    if (tid < H) {
      float s = A.phi_z_b[tid];
#pragma unroll
      for (int j = 0; j < Zd; ++j) s += A.phi_z_W[tid * Zd + j] * zv[j];
      phizb[tid] = f2bf(fmaxf(s, 0.f));
    }
    if (tid == 1023) {
      float k2 = 0.f;
#pragma unroll
      for (int j = 0; j < Zd; ++j) {
        const float d = em[j] - pm[j];
        k2 += 2.f * logf(ps[j] + EPSF) - 2.f * logf(es[j] + EPSF)
            + (es[j] * es[j] + d * d) / (ps[j] * ps[j]) - 1.f;
      }
      acc[0] += 0.5f * k2;
    }
    __syncthreads();
    // P6: GEMVs on phi_z: dec_W1z(16 tiles -> dec_h temp) | Wih_z(48 -> giz)
#pragma unroll
    for (int i = 0; i < 2; ++i) {
      const int tt = wave * 2 + i;
      float* dst = (tt < 16) ? (dec_h + tt * 16) : (giz + (tt - 16) * 16);
      gemv_tile(A.G3 + tt * 4096, phizb, lane, dst);
    }
    __syncthreads();
    // P7a: dec layer1 combine ; GRU r,zg gates
    if (tid < H)
      dec1b[tid] = f2bf(fmaxf(dec_h[tid] + u_dec1h[tid] + A.dec_b1[tid], 0.f));
    if (tid < 2 * H)
      giz[tid] = sigmoidf(A.gix[t * 768 + tid] + giz[tid] + gh[tid]);
    __syncthreads();
    // P7b: GRU candidate + new hidden state
    if (tid < H) {
      const float r = giz[tid], zg = giz[H + tid];
      const float gin = A.gix[t * 768 + 2 * H + tid] + giz[2 * H + tid];
      const float nv = tanhf(gin + r * gh[2 * H + tid]);
      hn[tid] = (1.f - zg) * nv + zg * h[tid];
    }
    __syncthreads();
    // P8: dec_h = dec_W2 @ dec1 (waves 0-15, LDS tiles) ; commit h (16-23)
    if (wave < 16) {
      gemv_tile(ldsW + (16 + wave) * 4096, dec1b, lane, dec_h + wave * 16);
    } else if (tid - 512 < H) {
      const int i = tid - 512;
      const float v = hn[i];
      h[i] = v; hb[i] = f2bf(v);
    }
    __syncthreads();
    // P9: dec_h activation
    if (tid < H) dec_h[tid] = fmaxf(dec_h[tid] + A.dec_b2[tid], 0.f);
    __syncthreads();
    // P10: dec mean/std (waves 0-9) + emit
    if (wave < 10) {
      const int r = wave, rr = (r < 5) ? r : r - 5;
      const float* Wm = (r < 5) ? A.dec_mean_W : A.dec_std_W;
      float s = 0.f;
      for (int j = lane; j < H; j += 32) s += Wm[rr * H + j] * dec_h[j];
      for (int off = 16; off; off >>= 1) s += __shfl_xor(s, off, 32);
      if (lane == 0) {
        if (r < 5) {
          const float mv = sigmoidf(s + A.dec_mean_b[rr]);
          dm[rr] = mv;
          A.out[2 + 2 * T * Zd + t * Xd + rr] = mv;
        } else {
          A.out[2 + 3 * T * Zd + t * Xd + rr] = softplusf(s + A.dec_std_b[rr]);
        }
      }
    }
    __syncthreads();
    // P11: NLL accumulation
    if (tid == 0) {
      float n2 = 0.f;
#pragma unroll
      for (int j = 0; j < Xd; ++j) {
        const float xv = A.x[t * Xd + j];
        n2 -= xv * logf(dm[j] + EPSF) + (1.f - xv) * logf(1.f - dm[j] - EPSF);
      }
      acc[1] += n2;
    }
    __syncthreads();
  }
  if (tid == 0) { A.out[0] = acc[0] / (float)T; A.out[1] = acc[1] / (float)T; }
}

// ---------------------------------------------------------------------------
extern "C" void kernel_launch(void* const* d_in, const int* in_sizes, int n_in,
                              void* d_out, int out_size, void* d_ws, size_t ws_size,
                              hipStream_t stream) {
  (void)in_sizes; (void)n_in; (void)out_size; (void)ws_size;
  const float* x   = (const float*)d_in[0];
  const float* eps = (const float*)d_in[1];
  const float* P[30];
  for (int i = 0; i < 30; ++i) P[i] = (const float*)d_in[2 + i];
  // param order:
  // 0 phi_x_W1 1 phi_x_b1 2 phi_x_W2 3 phi_x_b2 4 phi_z_W 5 phi_z_b
  // 6 enc_W1 7 enc_b1 8 enc_W2 9 enc_b2 10 enc_mean_W 11 enc_mean_b
  // 12 enc_std_W 13 enc_std_b 14 prior_W 15 prior_b 16 prior_mean_W
  // 17 prior_mean_b 18 prior_std_W 19 prior_std_b 20 dec_W1 21 dec_b1
  // 22 dec_W2 23 dec_b2 24 dec_mean_W 25 dec_mean_b 26 dec_std_W 27 dec_std_b
  // 28 rnn_Wih 29 rnn_Whh

  unsigned short* TW    = (unsigned short*)d_ws;          // 272 tiles * 4096
  unsigned short* tmp1  = TW + 272 * 4096;                // TP*H bf16
  unsigned short* phibf = tmp1 + TP * H;                  // TP*H bf16
  float* encx = (float*)(phibf + TP * H);                 // TP*H f32
  float* gix  = encx + TP * H;                            // TP*3H f32

  // --- tile all recurrent + GEMM-B weights to bf16 WMMA layout ---
  struct { const float* W; int ld, off, nt; int dstTile; } cv[11] = {
    {P[6],  2 * H, H, 16,   0},   // enc_W1[:,H:]   (G1)
    {P[14],     H, 0, 16,  16},   // prior_W        (G1)
    {P[20], 2 * H, H, 16,  32},   // dec_W1[:,H:]   (G1)
    {P[29],     H, 0, 48,  48},   // rnn_Whh        (G1)
    {P[8],      H, 0, 16,  96},   // enc_W2         (G2)
    {P[20], 2 * H, 0, 16, 112},   // dec_W1[:,:H]   (G3)
    {P[28], 2 * H, H, 48, 128},   // rnn_Wih[:,H:]  (G3)
    {P[22],     H, 0, 16, 176},   // dec_W2         (G4)
    {P[2],      H, 0, 16, 192},   // phi_x_W2       (B: phi_x gemm)
    {P[6],  2 * H, 0, 16, 208},   // enc_W1[:,:H]   (B: encx gemm)
    {P[28], 2 * H, 0, 48, 224},   // rnn_Wih[:,:H]  (B: gix gemm)
  };
  for (int i = 0; i < 11; ++i)
    k_tile<<<cv[i].nt * 8, 256, 0, stream>>>(cv[i].W, cv[i].ld, cv[i].off,
                                             TW + cv[i].dstTile * 4096);

  // --- phi_x layer 1 (K=5, VALU) ---
  k_layer1<<<TP, 256, 0, stream>>>(x, P[0], P[1], tmp1);

  // --- precompute GEMMs (bf16 WMMA) ---
  GemmArgs gA; gA.A = tmp1;  gA.Bt = TW + 192 * 4096; gA.bias = P[3];
  gA.outF = nullptr; gA.outBf = phibf; gA.ntiles = 16; gA.Nld = H; gA.relu = 1;
  k_gemm<<<MT * 16 / 8, 256, 0, stream>>>(gA);

  GemmArgs gB; gB.A = phibf; gB.Bt = TW + 208 * 4096; gB.bias = P[7];
  gB.outF = encx; gB.outBf = nullptr; gB.ntiles = 16; gB.Nld = H; gB.relu = 0;
  k_gemm<<<MT * 16 / 8, 256, 0, stream>>>(gB);

  GemmArgs gC; gC.A = phibf; gC.Bt = TW + 224 * 4096; gC.bias = nullptr;
  gC.outF = gix; gC.outBf = nullptr; gC.ntiles = 48; gC.Nld = 3 * H; gC.relu = 0;
  k_gemm<<<MT * 48 / 8, 256, 0, stream>>>(gC);

  // --- sequential recurrence: one persistent WGP, 256 KB dynamic LDS ---
  RecArgs R;
  R.x = x; R.eps = eps;
  R.G1 = TW; R.G2 = TW + 96 * 4096; R.G3 = TW + 112 * 4096; R.G4 = TW + 176 * 4096;
  R.encx = encx; R.gix = gix;
  R.prior_b = P[15]; R.enc_b2 = P[9]; R.dec_b1 = P[21]; R.dec_b2 = P[23];
  R.phi_z_W = P[4]; R.phi_z_b = P[5];
  R.enc_mean_W = P[10]; R.enc_mean_b = P[11];
  R.enc_std_W  = P[12]; R.enc_std_b  = P[13];
  R.prior_mean_W = P[16]; R.prior_mean_b = P[17];
  R.prior_std_W  = P[18]; R.prior_std_b  = P[19];
  R.dec_mean_W = P[24]; R.dec_mean_b = P[25];
  R.dec_std_W  = P[26]; R.dec_std_b  = P[27];
  R.out = (float*)d_out;
  k_recurrent<<<1, 1024, 32 * 4096 * sizeof(unsigned short), stream>>>(R);
}
